// LinearAttention_6408091206043
// MI455X (gfx1250) — compile-verified, matrix operand order checked by reading, not run on hardware
//
#include <hip/hip_runtime.h>
#include <hip/hip_bf16.h>
#include <math.h>

// ---------------------------------------------------------------------------
// LinearAttention for MI455X (gfx1250, wave32, WMMA + TDM).
// bf16 inputs + f32 accumulation via v_wmma_f32_16x16x32_bf16 for all four
// GEMMs; pointwise math (bias, RoPE, elu+1, normalizer) in f32.
// Kernel 4 stages its bf16 B-tiles with the Tensor Data Mover
// (tensor_load_to_lds + s_wait_tensorcnt), using the D# pad feature to insert
// the LDS row padding the WMMA fragment loader expects.
// ---------------------------------------------------------------------------

typedef __bf16 bf16_t;
typedef __attribute__((ext_vector_type(16))) __bf16 v16bf;
typedef __attribute__((ext_vector_type(8)))  __bf16 v8bf;
typedef __attribute__((ext_vector_type(8)))  float   v8f;
typedef __attribute__((ext_vector_type(4)))  unsigned int u32x4;
typedef __attribute__((ext_vector_type(8)))  int i32x8;
typedef __attribute__((ext_vector_type(4)))  int i32x4;

#define EMBED   384
#define NB      8
#define HWDIM   4096
#define NHEAD   6
#define HDIM    64
#define LOG_THETA 9.210340371976184f   // ln(10000)

// D = A*B + C, bf16 inputs, f32 accumulate (16x16x32)
__device__ __forceinline__ v8f wmma_bf16(v16bf a, v16bf b, v8f c) {
    return __builtin_amdgcn_wmma_f32_16x16x32_bf16(false, a, false, b,
                                                   (short)0, c, false, false);
}

// Build one 16-element bf16 fragment for a lane.
// Per ISA 16-bit A/B layout: lane row holds K = {koff..koff+7, koff+16..koff+23}
// where koff = 0 for lanes 0-15 and 8 for lanes 16-31 (two 16B chunks).
__device__ __forceinline__ v16bf frag16(const bf16_t* p, int koff) {
    v8bf lo = *(const v8bf*)(p + koff);
    v8bf hi = *(const v8bf*)(p + koff + 16);
    return __builtin_shufflevector(lo, hi, 0,1,2,3,4,5,6,7,
                                           8,9,10,11,12,13,14,15);
}

// Pair-strip channel map: block group g covers 64 channels arranged as the two
// RoPE rotation partners {i, i+96} (within each 192-channel half).
__device__ __forceinline__ int ch_of(int g, int r) {
    int base = (g < 3) ? g * 32 : 192 + (g - 3) * 32;
    return base + ((r < 32) ? r : 64 + r);
}

// ---------------------------------------------------------------------------
// TDM: DMA a 2-D bf16 tile (tile_d1 rows x tile_d0 contiguous elements) from
// global memory into LDS at byte offset lds_off, inserting pad_amount dwords
// of padding every pad_interval dwords (codes per D# group1 encoding).
// D# bitfields per CDNA5 ISA ch.8 (group0: count/lds_addr/global_addr/type,
// group1: data_size/pad/tensor dims/tile dims/stride).
// ---------------------------------------------------------------------------
__device__ __forceinline__ void tdm_load_2d_bf16(
    unsigned lds_off, const void* gaddr,
    unsigned tensor_d0, unsigned tensor_d1,
    unsigned tile_d0, unsigned tile_d1,
    unsigned stride_d0,
    unsigned pad_interval_code, unsigned pad_amount_code)
{
    unsigned long long ga = (unsigned long long)(size_t)gaddr;
    u32x4 g0;
    g0[0] = 1u;                                         // count=1, user D#
    g0[1] = lds_off;                                    // lds_addr (bytes)
    g0[2] = (unsigned)(ga & 0xFFFFFFFFu);               // global_addr[31:0]
    g0[3] = (unsigned)((ga >> 32) & 0x1FFFFFFu)         // global_addr[56:32]
            | (2u << 30);                               // type=2 ("image")
    i32x8 g1;
    g1[0] = (int)((1u << 16)                            // data_size = 2 bytes
            | (1u << 20)                                // pad_enable
            | (pad_interval_code << 22)
            | (pad_amount_code << 25));
    g1[1] = (int)((tensor_d0 & 0xFFFFu) << 16);         // tensor_dim0[15:0] @63:48
    g1[2] = (int)(((tensor_d0 >> 16) & 0xFFFFu)         // tensor_dim0[31:16]
            | ((tensor_d1 & 0xFFFFu) << 16));           // tensor_dim1[15:0]
    g1[3] = (int)(((tensor_d1 >> 16) & 0xFFFFu)         // tensor_dim1[31:16]
            | ((tile_d0 & 0xFFFFu) << 16));             // tile_dim0 @127:112
    g1[4] = (int)(tile_d1 & 0xFFFFu);                   // tile_dim1; tile_dim2=0
    g1[5] = (int)stride_d0;                             // tensor_dim0_stride lo32
    g1[6] = 0;                                          // stride hi16, dim1_stride
    g1[7] = 0;
    i32x4 z4 = {0, 0, 0, 0};
#if defined(__clang_major__) && __clang_major__ >= 23
    i32x8 z8 = {0, 0, 0, 0, 0, 0, 0, 0};
    __builtin_amdgcn_tensor_load_to_lds(g0, g1, z4, z4, z8, 0);
#else
    __builtin_amdgcn_tensor_load_to_lds(g0, g1, z4, z4, 0);
#endif
}

// ---------------------------------------------------------------------------
// Kernel 1: fused QKV projection GEMM + bias + RoPE + elu+1, bf16 outputs.
// grid (HW/128, 6 pair-groups, B*3), block 256 (8 waves, 2x4 wave grid).
// Tile: M=64 (pair strips) x N=128, K-loop over 384 in steps of 32.
// ---------------------------------------------------------------------------
__global__ __launch_bounds__(256) void qkv_rope_kernel(
    const float* __restrict__ x,
    const float* __restrict__ Wq, const float* __restrict__ bq,
    const float* __restrict__ Wk, const float* __restrict__ bk,
    const float* __restrict__ Wv, const float* __restrict__ bv,
    bf16_t* __restrict__ qh, bf16_t* __restrict__ kT, bf16_t* __restrict__ vT)
{
    __shared__ alignas(16) bf16_t lds_a[64 * 40];     // W tile  [row][k]
    __shared__ alignas(16) bf16_t lds_b[128 * 40];    // x tile  [n][k]
    __shared__ float lds_o[64 * 132];                 // f32 result staging

    const int t    = threadIdx.x;
    const int lane = t & 31;
    const int wv   = t >> 5;
    const int wm   = wv >> 2;            // 0..1  (M strip of 32)
    const int wn   = wv & 3;             // 0..3  (N strip of 32)
    const int n0   = blockIdx.x * 128;
    const int g    = blockIdx.y;
    const int b    = blockIdx.z / 3;
    const int mat  = blockIdx.z % 3;     // 0=q 1=k 2=v

    const float* Wsel = (mat == 0) ? Wq : (mat == 1) ? Wk : Wv;
    const float* bsel = (mat == 0) ? bq : (mat == 1) ? bk : bv;

    const int koff = (lane >> 4) ? 8 : 0;
    const int lrow = lane & 15;
    v8f acc[2][2] = {};

    for (int kt = 0; kt < EMBED; kt += 32) {
        // --- stage A tile: 64 rows x 32 k of W (f32 -> bf16) ---
        {
            const int r  = t >> 2;
            const int kc = (t & 3) * 8;
            const float* src = Wsel + (size_t)ch_of(g, r) * EMBED + kt + kc;
            float4 f0 = *(const float4*)(src);
            float4 f1 = *(const float4*)(src + 4);
            bf16_t* dst = lds_a + r * 40 + kc;
            dst[0] = (bf16_t)f0.x; dst[1] = (bf16_t)f0.y;
            dst[2] = (bf16_t)f0.z; dst[3] = (bf16_t)f0.w;
            dst[4] = (bf16_t)f1.x; dst[5] = (bf16_t)f1.y;
            dst[6] = (bf16_t)f1.z; dst[7] = (bf16_t)f1.w;
        }
        // --- stage B tile: x[b][kt+kr][n0..n0+127], transposed to [n][k] ---
        {
            const int kr = t >> 3;
            const int nb = (t & 7) * 16;
            const float* src = x + ((size_t)b * EMBED + kt + kr) * HWDIM + n0 + nb;
            #pragma unroll
            for (int j4 = 0; j4 < 4; ++j4) {
                float4 f = *(const float4*)(src + j4 * 4);
                lds_b[(nb + j4 * 4 + 0) * 40 + kr] = (bf16_t)f.x;
                lds_b[(nb + j4 * 4 + 1) * 40 + kr] = (bf16_t)f.y;
                lds_b[(nb + j4 * 4 + 2) * 40 + kr] = (bf16_t)f.z;
                lds_b[(nb + j4 * 4 + 3) * 40 + kr] = (bf16_t)f.w;
            }
            if (kt + 32 < EMBED)
                __builtin_prefetch(src + (size_t)32 * HWDIM, 0, 1);
        }
        __syncthreads();

        v16bf afr0 = frag16(lds_a + (wm * 32 +      lrow) * 40, koff);
        v16bf afr1 = frag16(lds_a + (wm * 32 + 16 + lrow) * 40, koff);
        v16bf bfr0 = frag16(lds_b + (wn * 32 +      lrow) * 40, koff);
        v16bf bfr1 = frag16(lds_b + (wn * 32 + 16 + lrow) * 40, koff);
        acc[0][0] = wmma_bf16(afr0, bfr0, acc[0][0]);
        acc[0][1] = wmma_bf16(afr0, bfr1, acc[0][1]);
        acc[1][0] = wmma_bf16(afr1, bfr0, acc[1][0]);
        acc[1][1] = wmma_bf16(afr1, bfr1, acc[1][1]);
        __syncthreads();
    }

    // spill accumulators to LDS f32 so RoPE partners can be exchanged
    const int mrow = (lane >> 4) * 8;
    #pragma unroll
    for (int ti = 0; ti < 2; ++ti)
        #pragma unroll
        for (int tj = 0; tj < 2; ++tj)
            #pragma unroll
            for (int r = 0; r < 8; ++r)
                lds_o[(wm * 32 + ti * 16 + mrow + r) * 132 +
                      wn * 32 + tj * 16 + lrow] = acc[ti][tj][r];
    __syncthreads();

    // epilogue: bias + RoPE + elu+1, write head layouts
    const int rp  = t >> 3;            // 0..31: partner pair (rp, rp+32)
    const int nb  = (t & 7) * 16;
    const int ch1 = ch_of(g, rp);
    const int ch2 = ch_of(g, rp + 32);
    const float bia1 = bsel[ch1];
    const float bia2 = bsel[ch2];

    if (mat == 2) {   // v: bias only, write transposed [b,h,d,n]
        bf16_t* d1 = vT + ((size_t)(b * NHEAD + ch1 / HDIM) * HDIM + ch1 % HDIM) * HWDIM + n0 + nb;
        bf16_t* d2 = vT + ((size_t)(b * NHEAD + ch2 / HDIM) * HDIM + ch2 % HDIM) * HWDIM + n0 + nb;
        #pragma unroll
        for (int j = 0; j < 16; ++j) {
            d1[j] = (bf16_t)(lds_o[rp * 132 + nb + j] + bia1);
            d2[j] = (bf16_t)(lds_o[(rp + 32) * 132 + nb + j] + bia2);
        }
    } else {
        const int  half_off = (g < 3) ? 0 : 192;
        const bool posH     = (g < 3);
        const int  i1 = ch1 - half_off;        // 0..95   (first rot half)
        const int  i2 = ch2 - half_off;        // 96..191 (second rot half)
        const float inv1 = __expf(-LOG_THETA * (float)(i1 >> 1) / 96.f);
        const float inv2 = __expf(-LOG_THETA * (float)(i2 >> 1) / 96.f);
        #pragma unroll 4
        for (int j = 0; j < 16; ++j) {
            const int ng = n0 + nb + j;
            float v1 = lds_o[rp * 132 + nb + j] + bia1;
            float v2 = lds_o[(rp + 32) * 132 + nb + j] + bia2;
            const float pos = posH ? (float)(ng >> 6) : (float)(ng & 63);
            float s1, c1, s2, c2;
            __sincosf(pos * inv1, &s1, &c1);
            __sincosf(pos * inv2, &s2, &c2);
            float o1 = v1 * c1 - v2 * s1;      // rot[i<96]  = -t[i+96]
            float o2 = v2 * c2 + v1 * s2;      // rot[i>=96] =  t[i-96]
            if (mat == 0) { o1 *= 0.125f; o2 *= 0.125f; }   // hd^-0.5
            o1 = (o1 > 0.f) ? o1 + 1.f : __expf(o1);        // elu(x)+1
            o2 = (o2 > 0.f) ? o2 + 1.f : __expf(o2);
            if (mat == 0) {                    // q: [b,h,n,d]
                qh[((size_t)(b * NHEAD + ch1 / HDIM) * HWDIM + ng) * HDIM + (ch1 % HDIM)] = (bf16_t)o1;
                qh[((size_t)(b * NHEAD + ch2 / HDIM) * HWDIM + ng) * HDIM + (ch2 % HDIM)] = (bf16_t)o2;
            } else {                           // k: transposed [b,h,d,n]
                kT[((size_t)(b * NHEAD + ch1 / HDIM) * HDIM + ch1 % HDIM) * HWDIM + ng] = (bf16_t)o1;
                kT[((size_t)(b * NHEAD + ch2 / HDIM) * HDIM + ch2 % HDIM) * HWDIM + ng] = (bf16_t)o2;
            }
        }
    }
}

// ---------------------------------------------------------------------------
// Kernel 2: kv[b,h] = k^T v  (64 x 4096 x 64) and k_sum, 8-way split over n
// with f32 atomics. grid (8, NH, B), block 128 (4 waves, one 16-row strip ea.)
// ---------------------------------------------------------------------------
__global__ __launch_bounds__(128) void kv_ksum_kernel(
    const bf16_t* __restrict__ kT, const bf16_t* __restrict__ vT,
    float* __restrict__ kv, float* __restrict__ ksum)
{
    const int t = threadIdx.x, lane = t & 31, wv = t >> 5;
    const int n0 = blockIdx.x * 512;
    const size_t bh = (size_t)(blockIdx.z * NHEAD + blockIdx.y);
    const bf16_t* kb = kT + bh * HDIM * HWDIM;
    const bf16_t* vb = vT + bh * HDIM * HWDIM;
    const int koff = (lane >> 4) ? 8 : 0;
    const int lrow = lane & 15;

    v8f acc[4] = {};
    float ks_acc = 0.f;
    const bf16_t* arow = kb + (size_t)(wv * 16 + lrow) * HWDIM;

    for (int nt = 0; nt < 512; nt += 32) {
        v16bf afr = frag16(arow + n0 + nt, koff);
        #pragma unroll
        for (int i = 0; i < 16; ++i) ks_acc += (float)afr[i];
        #pragma unroll
        for (int tj = 0; tj < 4; ++tj) {
            v16bf bfr = frag16(vb + (size_t)(tj * 16 + lrow) * HWDIM + n0 + nt, koff);
            acc[tj] = wmma_bf16(afr, bfr, acc[tj]);
        }
    }

    // k_sum: lanes L and L^16 cover the same d-row -> fold then atomic add
    float tot = ks_acc + __shfl_xor(ks_acc, 16, 32);
    if (lane < 16) atomicAdd(&ksum[bh * HDIM + wv * 16 + lane], tot);

    const int mrow = (lane >> 4) * 8;
    #pragma unroll
    for (int tj = 0; tj < 4; ++tj)
        #pragma unroll
        for (int r = 0; r < 8; ++r)
            atomicAdd(&kv[(bh * HDIM + wv * 16 + mrow + r) * HDIM + tj * 16 + lrow],
                      acc[tj][r]);
}

// ---------------------------------------------------------------------------
// Kernel 3: out = (q @ kv) / (q . k_sum + 1e-6).
// Written bf16 POSITION-major [b][n][c] so kernel 4 can TDM-stage its tiles.
// grid (HW/128, NH, B), block 256 (8 waves, 16 n-rows each).
// ---------------------------------------------------------------------------
__global__ __launch_bounds__(256) void attn_out_kernel(
    const bf16_t* __restrict__ qh, const float* __restrict__ kv,
    const float* __restrict__ ksum, bf16_t* __restrict__ attn)
{
    __shared__ alignas(16) bf16_t lds_kv[64 * 72];   // kv^T [e][d] bf16
    __shared__ float lds_ks[64];
    __shared__ float lds_norm[128];

    const int t = threadIdx.x, lane = t & 31, wv = t >> 5;
    const int n0 = blockIdx.x * 128;
    const int h  = blockIdx.y, b = blockIdx.z;
    const size_t bh = (size_t)(b * NHEAD + h);
    const float* kvp = kv + bh * HDIM * HDIM;
    {
        const int d = t >> 2, ec = (t & 3) * 16;
        #pragma unroll
        for (int j = 0; j < 16; ++j)
            lds_kv[(ec + j) * 72 + d] = (bf16_t)kvp[d * HDIM + ec + j];
        if (t < 64) lds_ks[t] = ksum[bh * HDIM + t];
    }
    __syncthreads();

    const int koff = (lane >> 4) ? 8 : 0;
    const int lrow = lane & 15;
    const bf16_t* qrow = qh + (bh * HWDIM + n0 + wv * 16 + lrow) * HDIM;

    v8f acc[4] = {};
    float npart = 0.f;
    #pragma unroll
    for (int dt = 0; dt < HDIM; dt += 32) {
        v16bf afr = frag16(qrow + dt, koff);
        #pragma unroll
        for (int i = 0; i < 8; ++i) {
            npart += (float)afr[i]     * lds_ks[dt + koff + i];
            npart += (float)afr[i + 8] * lds_ks[dt + koff + 16 + i];
        }
        #pragma unroll
        for (int tj = 0; tj < 4; ++tj) {
            v16bf bfr = frag16(lds_kv + (tj * 16 + lrow) * 72 + dt, koff);
            acc[tj] = wmma_bf16(afr, bfr, acc[tj]);
        }
    }

    float norm = npart + __shfl_xor(npart, 16, 32);
    if (lane < 16) lds_norm[wv * 16 + lane] = norm;
    __syncthreads();

    const int mrow = (lane >> 4) * 8;
    #pragma unroll
    for (int tj = 0; tj < 4; ++tj) {
        #pragma unroll
        for (int r = 0; r < 8; ++r) {
            const int m   = mrow + r;
            const float d = lds_norm[wv * 16 + m] + 1e-6f;
            const float v = acc[tj][r] / d;
            const int e   = tj * 16 + lrow;
            attn[((size_t)b * HWDIM + n0 + wv * 16 + m) * EMBED + h * HDIM + e] = (bf16_t)v;
        }
    }
}

// ---------------------------------------------------------------------------
// Kernel 4: final conv1x1: out = Wfc @ attn + bfc, f32 output.
// B tile (bf16, [n][c] in global) is staged by the Tensor Data Mover directly
// into LDS (pad feature -> 40-element row stride). Dynamic LDS so the TDM
// lds_addr (byte offset 0) is known. grid (HW/128, C/64, B), block 256.
// ---------------------------------------------------------------------------
#define FC_LDSB_BYTES (128 * 40 * 2)    // B tile [128][40] bf16 at offset 0
#define FC_LDSA_BYTES (64 * 40 * 2)     // A tile [64][40] bf16

__global__ __launch_bounds__(256) void fc_kernel(
    const bf16_t* __restrict__ attn, const float* __restrict__ Wfc,
    const float* __restrict__ bfc, float* __restrict__ out)
{
    extern __shared__ char smem[];
    bf16_t* lds_b = (bf16_t*)smem;                     // offset 0 (TDM dest)
    bf16_t* lds_a = (bf16_t*)(smem + FC_LDSB_BYTES);

    const int t = threadIdx.x, lane = t & 31, wv = t >> 5;
    const int wm = wv >> 2, wn = wv & 3;
    const int n0 = blockIdx.x * 128;
    const int m0 = blockIdx.y * 64;
    const int b  = blockIdx.z;
    const int koff = (lane >> 4) ? 8 : 0;
    const int lrow = lane & 15;

    const bf16_t* tile_base = attn + ((size_t)b * HWDIM + n0) * EMBED;
    v8f acc[2][2] = {};

    for (int kt = 0; kt < EMBED; kt += 32) {
        // --- TDM: DMA 128 rows x 32 bf16 (row stride EMBED) into lds_b with
        //     4 dwords of pad every 16 dwords -> LDS row stride 40 elements.
        if (wv == 0)
            tdm_load_2d_bf16(/*lds_off=*/0u, tile_base + kt,
                             /*tensor_d0=*/EMBED, /*tensor_d1=*/HWDIM,
                             /*tile_d0=*/32, /*tile_d1=*/128,
                             /*stride_d0=*/EMBED,
                             /*pad_interval 16dw=*/3u, /*pad 4dw=*/3u);
        // --- stage A tile: 64 rows x 32 k of Wfc (f32 -> bf16), all threads
        {
            const int r = t >> 2, kc = (t & 3) * 8;
            const float* src = Wfc + (size_t)(m0 + r) * EMBED + kt + kc;
            float4 f0 = *(const float4*)src;
            float4 f1 = *(const float4*)(src + 4);
            bf16_t* dst = lds_a + r * 40 + kc;
            dst[0] = (bf16_t)f0.x; dst[1] = (bf16_t)f0.y;
            dst[2] = (bf16_t)f0.z; dst[3] = (bf16_t)f0.w;
            dst[4] = (bf16_t)f1.x; dst[5] = (bf16_t)f1.y;
            dst[6] = (bf16_t)f1.z; dst[7] = (bf16_t)f1.w;
        }
        __builtin_amdgcn_s_wait_tensorcnt(0);   // no-op for waves 1..7
        __syncthreads();

        v16bf afr0 = frag16(lds_a + (wm * 32 +      lrow) * 40, koff);
        v16bf afr1 = frag16(lds_a + (wm * 32 + 16 + lrow) * 40, koff);
        v16bf bfr0 = frag16(lds_b + (wn * 32 +      lrow) * 40, koff);
        v16bf bfr1 = frag16(lds_b + (wn * 32 + 16 + lrow) * 40, koff);
        acc[0][0] = wmma_bf16(afr0, bfr0, acc[0][0]);
        acc[0][1] = wmma_bf16(afr0, bfr1, acc[0][1]);
        acc[1][0] = wmma_bf16(afr1, bfr0, acc[1][0]);
        acc[1][1] = wmma_bf16(afr1, bfr1, acc[1][1]);
        __syncthreads();
    }

    const int mrow = (lane >> 4) * 8;
    #pragma unroll
    for (int ti = 0; ti < 2; ++ti)
        #pragma unroll
        for (int tj = 0; tj < 2; ++tj)
            #pragma unroll
            for (int r = 0; r < 8; ++r) {
                const int m = wm * 32 + ti * 16 + mrow + r;
                const int n = wn * 32 + tj * 16 + lrow;
                out[((size_t)b * EMBED + m0 + m) * HWDIM + n0 + n] =
                    acc[ti][tj][r] + bfc[m0 + m];
            }
}

// ---------------------------------------------------------------------------
extern "C" void kernel_launch(void* const* d_in, const int* in_sizes, int n_in,
                              void* d_out, int out_size, void* d_ws, size_t ws_size,
                              hipStream_t stream) {
    (void)in_sizes; (void)n_in; (void)out_size; (void)ws_size;
    const float* x   = (const float*)d_in[0];
    const float* Wq  = (const float*)d_in[1];
    const float* bq  = (const float*)d_in[2];
    const float* Wk  = (const float*)d_in[3];
    const float* bk  = (const float*)d_in[4];
    const float* Wv  = (const float*)d_in[5];
    const float* bv  = (const float*)d_in[6];
    const float* Wfc = (const float*)d_in[7];
    const float* bfc = (const float*)d_in[8];
    float* out = (float*)d_out;

    const size_t SQ  = (size_t)NB * NHEAD * HWDIM * HDIM * sizeof(bf16_t); // 25.2 MB
    const size_t KVB = (size_t)NB * NHEAD * HDIM * HDIM * sizeof(float);
    const size_t KSB = (size_t)NB * NHEAD * HDIM * sizeof(float);
    char* ws = (char*)d_ws;
    bf16_t* qh   = (bf16_t*)(ws);
    bf16_t* kT   = (bf16_t*)(ws + SQ);
    bf16_t* vT   = (bf16_t*)(ws + 2 * SQ);
    bf16_t* attn = (bf16_t*)(ws + 3 * SQ);     // [b][n][c] position-major
    float*  kv   = (float*)(ws + 4 * SQ);
    float*  ksum = (float*)(ws + 4 * SQ + KVB);

    hipMemsetAsync(ws + 4 * SQ, 0, KVB + KSB, stream);   // atomics need zeros

    qkv_rope_kernel<<<dim3(HWDIM / 128, 6, NB * 3), 256, 0, stream>>>(
        x, Wq, bq, Wk, bk, Wv, bv, qh, kT, vT);
    kv_ksum_kernel<<<dim3(HWDIM / 512, NHEAD, NB), 128, 0, stream>>>(
        kT, vT, kv, ksum);
    attn_out_kernel<<<dim3(HWDIM / 128, NHEAD, NB), 256, 0, stream>>>(
        qh, kv, ksum, attn);
    fc_kernel<<<dim3(HWDIM / 128, EMBED / 64, NB), 256,
                FC_LDSB_BYTES + FC_LDSA_BYTES, stream>>>(
        attn, Wfc, bfc, out);
}
